// SelfAttention_76312978915618
// MI455X (gfx1250) — compile-verified, matrix operand order checked by reading, not run on hardware
//
#include <hip/hip_runtime.h>

// ---------------------------------------------------------------------------
// Self-attention, N=8192, D=1024, fp32 in/out.
//   k1: Q = X Wq, K = X Wk, Vt = (X Wv)^T   (all f16 in workspace, 48 MB)
//   k2: flash-attention  out = softmax(Q K^T) V   (f32 accum, f16 WMMA,
//       double-buffered async-to-LDS staging of Q/K chunks)
// ---------------------------------------------------------------------------

#define NTOK 8192
#define DIM  1024

typedef _Float16 v16h __attribute__((ext_vector_type(16)));
typedef _Float16 v8h  __attribute__((ext_vector_type(8)));
typedef float    v8f  __attribute__((ext_vector_type(8)));
typedef int      v4i  __attribute__((ext_vector_type(4)));

#define AS1 __attribute__((address_space(1)))
#define AS3 __attribute__((address_space(3)))

#if defined(__has_builtin)
#if __has_builtin(__builtin_amdgcn_global_load_async_to_lds_b128) && \
    __has_builtin(__builtin_amdgcn_s_wait_asynccnt)
#define HAVE_ASYNC_LDS 1
#endif
#endif

// Copy 16B global -> LDS, async (ASYNCcnt) when available.
__device__ __forceinline__ void cp16_g2l(const _Float16* g, _Float16* l) {
#ifdef HAVE_ASYNC_LDS
  __builtin_amdgcn_global_load_async_to_lds_b128((AS1 v4i*)g, (AS3 v4i*)l,
                                                 0, 0);
#else
  *(v8h*)l = *(const v8h*)g;
#endif
}

// WMMA f16 A/B fragment (16x32) loader from a row-major f16 matrix.
// Per ISA layout: lane l -> row (l&15), k-base 8*(l>>4);
// vector elems 0..7 = k..k+7, elems 8..15 = k+16..k+23.
__device__ __forceinline__ v16h frag_load(const _Float16* base, int row0,
                                          int k0, int ld) {
  int lane = threadIdx.x & 31;
  int m = lane & 15, h = lane >> 4;
  const _Float16* p = base + (size_t)(row0 + m) * ld + k0 + h * 8;
  v8h lo = *(const v8h*)p;
  v8h hi = *(const v8h*)(p + 16);
  v16h r;
#pragma unroll
  for (int i = 0; i < 8; ++i) { r[i] = lo[i]; r[i + 8] = hi[i]; }
  return r;
}

__device__ __forceinline__ v8f wmma_f16(v16h a, v16h b, v8f c) {
  return __builtin_amdgcn_wmma_f32_16x16x32_f16(false, a, false, b,
                                                (short)0, c, false, false);
}

// ---------------------------------------------------------------------------
// Kernel 1: fused QKV projection.  grid = (N/128, 24): y/8 picks {Wq,Wk,Wv},
// y%8 picks 128-col slab.  8 waves; wave w computes a 16x128 output stripe.
// ---------------------------------------------------------------------------
__global__ __launch_bounds__(256) void qkv_gemm(
    const float* __restrict__ X, const float* __restrict__ Wq,
    const float* __restrict__ Wk, const float* __restrict__ Wv,
    _Float16* __restrict__ Qh, _Float16* __restrict__ Kh,
    _Float16* __restrict__ Vth) {
  __shared__ __align__(16) _Float16 Wt[128][40];  // W tile transposed

  const int tid  = threadIdx.x;
  const int wave = tid >> 5, lane = tid & 31;
  const int nl = lane & 15, h = lane >> 4;
  const int mrow0 = blockIdx.x * 128 + wave * 16;
  const int mat   = blockIdx.y >> 3;
  const int col0  = (blockIdx.y & 7) * 128;
  const float* W = (mat == 0) ? Wq : (mat == 1) ? Wk : Wv;

  v8f acc[8];
#pragma unroll
  for (int j = 0; j < 8; ++j)
#pragma unroll
    for (int r = 0; r < 8; ++r) acc[j][r] = 0.0f;

  for (int k0 = 0; k0 < DIM; k0 += 32) {
#pragma unroll
    for (int i = 0; i < 16; ++i) {
      int idx = tid + i * 256;  // 0..4095
      int kk = idx >> 7;        // 0..31
      int nn = idx & 127;       // 0..127
      Wt[nn][kk] = (_Float16)W[(size_t)(k0 + kk) * DIM + col0 + nn];
    }
    __syncthreads();

    const float* px = X + (size_t)(mrow0 + nl) * DIM + k0 + h * 8;
    v16h a;
#pragma unroll
    for (int i = 0; i < 8; ++i) {
      a[i]     = (_Float16)px[i];
      a[i + 8] = (_Float16)px[i + 16];
    }

#pragma unroll
    for (int j = 0; j < 8; ++j) {
      const _Float16* pb = &Wt[j * 16 + nl][h * 8];
      v8h lo = *(const v8h*)pb;
      v8h hi = *(const v8h*)(pb + 16);
      v16h b;
#pragma unroll
      for (int i = 0; i < 8; ++i) { b[i] = lo[i]; b[i + 8] = hi[i]; }
      acc[j] = wmma_f16(a, b, acc[j]);
    }
    __syncthreads();
  }

  if (mat < 2) {
    _Float16* dst = (mat == 0) ? Qh : Kh;
#pragma unroll
    for (int j = 0; j < 8; ++j)
#pragma unroll
      for (int r = 0; r < 8; ++r)
        dst[(size_t)(mrow0 + r + 8 * h) * DIM + col0 + j * 16 + nl] =
            (_Float16)acc[j][r];
  } else {  // V stored transposed: Vt[d][token]
#pragma unroll
    for (int j = 0; j < 8; ++j)
#pragma unroll
      for (int r = 0; r < 8; ++r)
        Vth[(size_t)(col0 + j * 16 + nl) * NTOK + mrow0 + r + 8 * h] =
            (_Float16)acc[j][r];
  }
}

// ---------------------------------------------------------------------------
// Kernel 2: flash attention.  One workgroup = 64 query rows, 16 waves (512).
//   S role : wave w -> mtile = w&3 (16 rows), cols [32*(w>>2), +32)
//   O role : wave w -> output d-chunk [64*w, +64)  (128 VGPR accumulator)
// Q/K chunks double-buffered in LDS via async-to-LDS (issue-ahead with
// s_wait_asynccnt 6); P f16 in LDS; online softmax stats in LDS.
// ---------------------------------------------------------------------------

// Issue the 6 async 16B copies (2 Q rows-worth + 4 K rows-worth per thread)
// for one 128-wide k chunk.
__device__ __forceinline__ void stage_qk(const _Float16* __restrict__ Qh,
                                         const _Float16* __restrict__ Kh,
                                         _Float16 (*Qc)[136],
                                         _Float16 (*Ksh)[136], int qrow0,
                                         int key0, int kb, int tid) {
#pragma unroll
  for (int i = 0; i < 2; ++i) {
    int e = (tid + i * 512) * 8;  // 0..8191
    int r = e >> 7, c = e & 127;
    cp16_g2l(Qh + (size_t)(qrow0 + r) * DIM + kb + c, &Qc[r][c]);
  }
#pragma unroll
  for (int i = 0; i < 4; ++i) {
    int e = (tid + i * 512) * 8;  // 0..16383
    int r = e >> 7, c = e & 127;
    cp16_g2l(Kh + (size_t)(key0 + r) * DIM + kb + c, &Ksh[r][c]);
  }
}

__global__ __launch_bounds__(512) void flash_attn(
    const _Float16* __restrict__ Qh, const _Float16* __restrict__ Kh,
    const _Float16* __restrict__ Vth, float* __restrict__ Out) {
  __shared__ __align__(16) _Float16 Qc[2][64][136];    // Q chunk  (64 x 128) x2
  __shared__ __align__(16) _Float16 Ksh[2][128][136];  // K chunk (128 x 128) x2
  __shared__ __align__(16) _Float16 Pl[64][136];       // P tile   (64 x 128)
  __shared__ float pmax[64][4];
  __shared__ float psum[64][4];
  __shared__ float run_max[64], run_sum[64], row_scale[64], row_nm[64];

  const int tid  = threadIdx.x;
  const int wave = tid >> 5, lane = tid & 31;
  const int n = lane & 15, h = lane >> 4;
  const int qrow0 = blockIdx.x * 64;
  const int mtile = wave & 3;   // S-compute row tile
  const int npair = wave >> 2;  // S-compute column pair (0..3)
  const int d0    = wave * 64;  // output d-chunk

  if (tid < 64) { run_max[tid] = -1e30f; run_sum[tid] = 0.0f; }

  v8f O[4][4];
#pragma unroll
  for (int mt = 0; mt < 4; ++mt)
#pragma unroll
    for (int j = 0; j < 4; ++j)
#pragma unroll
      for (int r = 0; r < 8; ++r) O[mt][j][r] = 0.0f;
  __syncthreads();

  for (int key0 = 0; key0 < NTOK; key0 += 128) {
    // ---- S = Q K^T, k blocked by 128, double-buffered async staging ----
    v8f S0, S1;
#pragma unroll
    for (int r = 0; r < 8; ++r) { S0[r] = 0.0f; S1[r] = 0.0f; }

    stage_qk(Qh, Kh, Qc[0], Ksh[0], qrow0, key0, 0, tid);
    for (int c = 0; c < 8; ++c) {
      const int cur = c & 1;
      if (c < 7) {
        // issue-ahead: start next chunk, then wait only for current's 6 loads
        stage_qk(Qh, Kh, Qc[cur ^ 1], Ksh[cur ^ 1], qrow0, key0,
                 (c + 1) * 128, tid);
#ifdef HAVE_ASYNC_LDS
        __builtin_amdgcn_s_wait_asynccnt(6);
#endif
      } else {
#ifdef HAVE_ASYNC_LDS
        __builtin_amdgcn_s_wait_asynccnt(0);
#endif
      }
      __syncthreads();  // current buffer visible to all waves

#pragma unroll
      for (int k = 0; k < 128; k += 32) {
        v16h a  = frag_load(&Qc[cur][0][0], mtile * 16, k, 136);
        v16h b0 = frag_load(&Ksh[cur][0][0], npair * 32, k, 136);
        v16h b1 = frag_load(&Ksh[cur][0][0], npair * 32 + 16, k, 136);
        S0 = wmma_f16(a, b0, S0);
        S1 = wmma_f16(a, b1, S1);
      }
      __syncthreads();  // all waves done reading 'cur' before it is re-staged
    }

    // ---- per-row block max (reduce across 16-lane column group) ----
#pragma unroll
    for (int r = 0; r < 8; ++r) {
      float v = fmaxf(S0[r], S1[r]);
      v = fmaxf(v, __shfl_xor(v, 1, 32));
      v = fmaxf(v, __shfl_xor(v, 2, 32));
      v = fmaxf(v, __shfl_xor(v, 4, 32));
      v = fmaxf(v, __shfl_xor(v, 8, 32));
      if (n == 0) pmax[mtile * 16 + r + 8 * h][npair] = v;
    }
    __syncthreads();

    if (tid < 64) {
      float bm = fmaxf(fmaxf(pmax[tid][0], pmax[tid][1]),
                       fmaxf(pmax[tid][2], pmax[tid][3]));
      float om = run_max[tid];
      float nm = fmaxf(om, bm);
      run_max[tid]   = nm;
      row_nm[tid]    = nm;
      row_scale[tid] = __expf(om - nm);
    }
    __syncthreads();

    // ---- P = exp(S - max) -> LDS(f16); partial row sums ----
#pragma unroll
    for (int r = 0; r < 8; ++r) {
      int row = mtile * 16 + r + 8 * h;
      float nm = row_nm[row];
      float p0 = __expf(S0[r] - nm);
      float p1 = __expf(S1[r] - nm);
      Pl[row][npair * 32 + n]      = (_Float16)p0;
      Pl[row][npair * 32 + 16 + n] = (_Float16)p1;
      float s = p0 + p1;
      s += __shfl_xor(s, 1, 32);
      s += __shfl_xor(s, 2, 32);
      s += __shfl_xor(s, 4, 32);
      s += __shfl_xor(s, 8, 32);
      if (n == 0) psum[row][npair] = s;
    }

    // ---- rescale O accumulators by exp(old_max - new_max) ----
#pragma unroll
    for (int mt = 0; mt < 4; ++mt)
#pragma unroll
      for (int r = 0; r < 8; ++r) {
        float sc = row_scale[mt * 16 + r + 8 * h];
#pragma unroll
        for (int j = 0; j < 4; ++j) O[mt][j][r] *= sc;
      }
    __syncthreads();

    if (tid < 64)
      run_sum[tid] = run_sum[tid] * row_scale[tid] + psum[tid][0] +
                     psum[tid][1] + psum[tid][2] + psum[tid][3];

    // ---- O += P V for this wave's 64-wide d-chunk ----
#pragma unroll
    for (int kk = 0; kk < 128; kk += 32) {
      v16h a0 = frag_load(&Pl[0][0], 0, kk, 136);
      v16h a1 = frag_load(&Pl[0][0], 16, kk, 136);
      v16h a2 = frag_load(&Pl[0][0], 32, kk, 136);
      v16h a3 = frag_load(&Pl[0][0], 48, kk, 136);
#pragma unroll
      for (int j = 0; j < 4; ++j) {
        v16h b = frag_load(Vth, d0 + j * 16, key0 + kk, NTOK);
        O[0][j] = wmma_f16(a0, b, O[0][j]);
        O[1][j] = wmma_f16(a1, b, O[1][j]);
        O[2][j] = wmma_f16(a2, b, O[2][j]);
        O[3][j] = wmma_f16(a3, b, O[3][j]);
      }
    }
    __syncthreads();
  }

  // ---- finalize: divide by softmax denominator, store f32 ----
#pragma unroll
  for (int mt = 0; mt < 4; ++mt)
#pragma unroll
    for (int r = 0; r < 8; ++r) {
      float inv = 1.0f / run_sum[mt * 16 + r + 8 * h];
      int row = qrow0 + mt * 16 + r + 8 * h;
#pragma unroll
      for (int j = 0; j < 4; ++j)
        Out[(size_t)row * DIM + d0 + j * 16 + n] = O[mt][j][r] * inv;
    }
}

// ---------------------------------------------------------------------------
extern "C" void kernel_launch(void* const* d_in, const int* in_sizes, int n_in,
                              void* d_out, int out_size, void* d_ws,
                              size_t ws_size, hipStream_t stream) {
  (void)in_sizes; (void)n_in; (void)out_size; (void)ws_size;
  const float* X  = (const float*)d_in[0];
  const float* Wq = (const float*)d_in[1];
  const float* Wk = (const float*)d_in[2];
  const float* Wv = (const float*)d_in[3];
  float* Out = (float*)d_out;

  // workspace: Q (16MB) | K (16MB) | V^T (16MB), all f16
  _Float16* Qh  = (_Float16*)d_ws;
  _Float16* Kh  = Qh + (size_t)NTOK * DIM;
  _Float16* Vth = Kh + (size_t)NTOK * DIM;

  dim3 g1(NTOK / 128, 24);
  qkv_gemm<<<g1, 256, 0, stream>>>(X, Wq, Wk, Wv, Qh, Kh, Vth);
  flash_attn<<<NTOK / 64, 512, 0, stream>>>(Qh, Kh, Vth, Out);
}
